// AttentionPool_30520037605741
// MI455X (gfx1250) — compile-verified, matrix operand order checked by reading, not run on hardware
//
#include <hip/hip_runtime.h>

#define N_ATOMS   131072
#define FEAT      512
#define N_GRAPHS  4096
#define NEG_SLOPE 0.2f

#define GPB       4          // graphs per block
#define MROWS     128        // 4 graphs * 32 padded rows
#define A_STRIDE  1040       // LDS row stride in bytes: 512*2 + 16 pad (bank-conflict free)

typedef __attribute__((ext_vector_type(16))) __bf16 bf16x16;
typedef __attribute__((ext_vector_type(8)))  float  f32x8;

union Frag { uint4 u[2]; bf16x16 v; };

__device__ __forceinline__ unsigned int f2bf1(float f) {
    // round-to-nearest-even f32 -> bf16 (bits in [15:0])
    unsigned int u = __float_as_uint(f);
    return (u + 0x7FFFu + ((u >> 16) & 1u)) >> 16;
}

__device__ __forceinline__ bf16x16 load_frag(const char* p) {
    // ISA 16-bit A/B fragment: K 0..7 (or 8..15) at +0, K 16..23 (or 24..31) at +32
    Frag f;
    f.u[0] = *(const uint4*)(p);
    f.u[1] = *(const uint4*)(p + 32);
    return f.v;
}

// ---------------- prep kernels ----------------

__global__ void prep_w_kernel(const float* __restrict__ W, unsigned int* __restrict__ Wb) {
    int i = blockIdx.x * 256 + threadIdx.x;       // each thread: 4 floats -> 4 bf16
    float4 v = *(const float4*)(W + (size_t)i * 4);
    uint2 p;
    p.x = f2bf1(v.x) | (f2bf1(v.y) << 16);
    p.y = f2bf1(v.z) | (f2bf1(v.w) << 16);
    ((uint2*)Wb)[i] = p;
}

__global__ void prep_off_kernel(const int* __restrict__ ncnt, int* __restrict__ goff) {
    if (threadIdx.x == 0 && blockIdx.x == 0) {
        int s = 0;
        for (int g = 0; g < N_GRAPHS; ++g) { goff[g] = s; s += ncnt[g]; }
    }
}

// ---------------- fused attention-pool kernel ----------------

__global__ __launch_bounds__(512, 1) void attnpool_kernel(
    const float*          __restrict__ A,
    const int*            __restrict__ ncnt,
    const int*            __restrict__ goff,
    const unsigned short* __restrict__ Wb,
    const float*          __restrict__ att,
    float*                __restrict__ out)
{
    __shared__ __align__(16) char  sA[MROWS * A_STRIDE];   // 130 KB bf16 A tile
    __shared__ float s_scores[MROWS];
    __shared__ float s_w[MROWS];
    __shared__ __align__(16) float s_out[GPB * FEAT];

    const int t  = threadIdx.x;
    const int g0 = blockIdx.x * GPB;

    // ---- init LDS reduction buffers ----
    if (t < MROWS) { s_scores[t] = 0.f; s_w[t] = 0.f; }
    *(float4*)&s_out[t * 4] = make_float4(0.f, 0.f, 0.f, 0.f);  // 512 thr * 4 = 2048

    // ---- stage A: f32 -> bf16 into LDS, zero pad rows ----
    const int col4 = t & 127;                      // float4 column index (0..127)
    for (int it = 0; it < 32; ++it) {
        int r   = it * 4 + (t >> 7);               // row 0..127
        int gl  = r >> 5;
        int a   = r & 31;
        int g   = g0 + gl;
        int cnt = ncnt[g]; cnt = cnt > 32 ? 32 : cnt;
        float4 v = make_float4(0.f, 0.f, 0.f, 0.f);
        if (a < cnt)
            v = *(const float4*)(A + (size_t)(goff[g] + a) * FEAT + col4 * 4);
        uint2 p;
        p.x = f2bf1(v.x) | (f2bf1(v.y) << 16);
        p.y = f2bf1(v.z) | (f2bf1(v.w) << 16);
        *(uint2*)(sA + r * A_STRIDE + col4 * 8) = p;
    }
    __syncthreads();

    // ---- wave tiling: 16 waves = 8 N-groups x 2 M-groups, 64x64 C tile per wave ----
    const int wv    = t >> 5;
    const int lane  = t & 31;
    const int lrow  = lane & 15;       // row (A) / col (B) within tile
    const int lhalf = lane >> 4;       // K-half select
    const int nbase = (wv & 7) * 64;
    const int mbase = (wv >> 3) * 64;

    f32x8 C[4][4] = {};                // 16 accumulator tiles (128 VGPRs)

    const char* wb = (const char*)Wb;
    for (int k0 = 0; k0 < FEAT; k0 += 32) {
        bf16x16 b[4];
        #pragma unroll
        for (int j = 0; j < 4; ++j)
            b[j] = load_frag(wb + (size_t)(nbase + 16 * j + lrow) * (FEAT * 2)
                                + k0 * 2 + lhalf * 16);
        #pragma unroll
        for (int i = 0; i < 4; ++i) {
            bf16x16 a = load_frag(sA + (mbase + 16 * i + lrow) * A_STRIDE
                                     + k0 * 2 + lhalf * 16);
            #pragma unroll
            for (int j = 0; j < 4; ++j)
                C[i][j] = __builtin_amdgcn_wmma_f32_16x16x32_bf16(
                    false, a, false, b[j], (short)0, C[i][j], false, false);
        }
    }

    // ---- per-atom attention logits: score[m] = sum_n att[n] * h[m,n] ----
    float attv[4];
    #pragma unroll
    for (int j = 0; j < 4; ++j) attv[j] = att[nbase + 16 * j + lrow];

    #pragma unroll
    for (int i = 0; i < 4; ++i) {
        #pragma unroll
        for (int r = 0; r < 8; ++r) {
            float p = 0.f;
            #pragma unroll
            for (int j = 0; j < 4; ++j) p += attv[j] * C[i][j][r];
            atomicAdd(&s_scores[mbase + 16 * i + 8 * lhalf + r], p);
        }
    }
    __syncthreads();

    // ---- LeakyReLU + stable softmax per graph (32-atom segments) ----
    if (t < GPB) {
        int cnt = ncnt[g0 + t]; cnt = cnt > 32 ? 32 : cnt;
        int base = t * 32;
        float m = -3.4e38f;
        for (int a = 0; a < cnt; ++a) {
            float sc = s_scores[base + a];
            sc = sc >= 0.f ? sc : NEG_SLOPE * sc;
            m = fmaxf(m, sc);
        }
        float s = 0.f;
        for (int a = 0; a < cnt; ++a) {
            float sc = s_scores[base + a];
            sc = sc >= 0.f ? sc : NEG_SLOPE * sc;
            float e = __expf(sc - m);
            s_w[base + a] = e;                 // pad rows stay 0
            s += e;
        }
        float inv = (s > 0.f) ? 1.f / s : 0.f;
        for (int a = 0; a < cnt; ++a) s_w[base + a] *= inv;
    }
    __syncthreads();

    // ---- weighted segment-sum: out[g,n] = sum_m w[m] * h[m,n] ----
    #pragma unroll
    for (int j = 0; j < 4; ++j) {
        int n = nbase + 16 * j + lrow;
        #pragma unroll
        for (int gp = 0; gp < 2; ++gp) {       // 2 graphs per wave M-range
            float acc = 0.f;
            #pragma unroll
            for (int ii = 0; ii < 2; ++ii) {   // 2 M-tiles per graph
                int i    = gp * 2 + ii;
                int rowb = mbase + 16 * i + 8 * lhalf;
                #pragma unroll
                for (int r = 0; r < 8; ++r) acc += s_w[rowb + r] * C[i][j][r];
            }
            int g = (mbase >> 5) + gp;
            atomicAdd(&s_out[g * FEAT + n], acc);
        }
    }
    __syncthreads();

    // ---- coalesced store of 4 graph rows ----
    {
        int idx = t * 4;
        int g   = idx >> 9;
        int n   = idx & 511;
        *(float4*)(out + (size_t)(g0 + g) * FEAT + n) = *(const float4*)&s_out[idx];
    }
}

// ---------------- host launch ----------------

extern "C" void kernel_launch(void* const* d_in, const int* in_sizes, int n_in,
                              void* d_out, int out_size, void* d_ws, size_t ws_size,
                              hipStream_t stream) {
    const float* A    = (const float*)d_in[0];   // [131072, 512] f32
    const int*   ncnt = (const int*)  d_in[1];   // [4096] int
    const float* W    = (const float*)d_in[2];   // [512, 512] f32
    const float* att  = (const float*)d_in[3];   // [512] f32
    float*       out  = (float*)d_out;           // [4096, 512] f32

    unsigned short* Wb   = (unsigned short*)d_ws;                    // 512 KB bf16 W
    int*            goff = (int*)((char*)d_ws + (size_t)512 * 1024); // 16 KB offsets

    prep_w_kernel  <<<256, 256, 0, stream>>>(W, (unsigned int*)Wb);
    prep_off_kernel<<<1,   32,  0, stream>>>(ncnt, goff);
    attnpool_kernel<<<N_GRAPHS / GPB, 512, 0, stream>>>(A, ncnt, goff, Wb, att, out);
}
